// TimeSlicedAdaptiveGraphConvolution_40699110097147
// MI455X (gfx1250) — matrix operand order repaired and checked by reading, not compiled, not run on hardware
//
#include <hip/hip_runtime.h>
#include <hip/hip_bf16.h>

typedef __attribute__((ext_vector_type(16))) __bf16 v16bf;
typedef __attribute__((ext_vector_type(8)))  float  v8f;

#define NB      64
#define C_IN    64
#define C_OUT   64
#define T_DIM   300
#define V_DIM   25
#define INTER_C 16
#define NSUB    3
#define TT      8
#define PPAD    208            // 200 pixels padded to 13 N-tiles of 16
#define XSTR    72             // bf16 row stride for 64-wide rows (bank-friendly)
#define MAB     96             // stacked a_w|b_w rows
#define GK      192            // stacked K for final conv (3*64)
#define GSTR    196            // bf16 row stride for 192-wide rows
#define HW      (T_DIM * V_DIM) // 7500

__device__ __forceinline__ unsigned short f2bf(float f) {
  unsigned int x = __float_as_uint(f);
  x += 0x7fffu + ((x >> 16) & 1u);           // round-to-nearest-even
  return (unsigned short)(x >> 16);
}
__device__ __forceinline__ float bf2f(unsigned short h) {
  return __uint_as_float(((unsigned int)h) << 16);
}

// ---- CDNA5 async global->LDS copy (ASYNCcnt-tracked) ----
__device__ __forceinline__ void async_load_f32_to_lds(float* lds_dst, const float* gsrc) {
#if __has_builtin(__builtin_amdgcn_global_load_async_to_lds_b32)
  __builtin_amdgcn_global_load_async_to_lds_b32(
      (__attribute__((address_space(1))) int*)gsrc,
      (__attribute__((address_space(3))) int*)lds_dst, 0, 0);
#else
  unsigned int lds_off = (unsigned int)(unsigned long long)lds_dst; // low 32 bits of the pointer give the LDS address
  unsigned long long ga = (unsigned long long)gsrc;
  asm volatile("global_load_async_to_lds_b32 %0, %1, off"
               :: "v"(lds_off), "v"(ga) : "memory");
#endif
}
__device__ __forceinline__ void async_wait_all() {
#if __has_builtin(__builtin_amdgcn_s_wait_asynccnt)
  __builtin_amdgcn_s_wait_asynccnt(0);
#else
  asm volatile("s_wait_asynccnt 0x0" ::: "memory");
#endif
}

// 16-bit A fragment (16x32): lane L -> M = L&15; kb = (L&16)?8:0;
// regs 0..3 = K kb+0..7 ; regs 4..7 = K kb+16..23 (pairs)
__device__ __forceinline__ v16bf load_fragA(const unsigned short* base, int row,
                                            int k0, int stride, int lane) {
  union { v16bf v; unsigned int u[8]; } f;
  const unsigned short* rp = base + row * stride + k0 + ((lane & 16) ? 8 : 0);
#pragma unroll
  for (int j = 0; j < 4; ++j) {
    f.u[j]     = *(const unsigned int*)(rp + 2 * j);
    f.u[j + 4] = *(const unsigned int*)(rp + 16 + 2 * j);
  }
  return f.v;
}
// 16-bit B fragment (32x16): lane L -> N = L&15; kb = (L&16)?8:0; regs 0..3 = K kb+0..7 ; regs 4..7 = K kb+16..23
__device__ __forceinline__ v16bf load_fragB(const unsigned short* base, int col,
                                            int k0, int stride, int lane) {
  union { v16bf v; unsigned int u[8]; } f;
  const unsigned short* rp = base + col * stride + k0 + ((lane & 16) ? 8 : 0);
#pragma unroll
  for (int j = 0; j < 8; ++j)
    f.u[j] = *(const unsigned int*)(rp + 2 * j + ((j & 4) ? 8 : 0));
  return f.v;
}

__global__ __launch_bounds__(256, 1)
void agc_fused_kernel(const float* __restrict__ x,
                      const float* __restrict__ Amat,
                      const float* __restrict__ gattn,
                      const float* __restrict__ g_w,
                      const float* __restrict__ g_b,
                      const float* __restrict__ a_w,
                      const float* __restrict__ a_b,
                      const float* __restrict__ b_w,
                      const float* __restrict__ b_b,
                      const float* __restrict__ bn_gamma,
                      const float* __restrict__ bn_beta,
                      const float* __restrict__ bn_mean,
                      const float* __restrict__ bn_var,
                      float* __restrict__ out)
{
  __shared__ __align__(16) unsigned short s_xbf[PPAD * XSTR];   // x tile, bf16 [p][c]
  __shared__ __align__(16) float s_abxs[20384];  // union: f32 x-stage[64*208] / AB f32[96*208] / xs bf16[208*196]
  __shared__ float s_scale[NSUB * PPAD];
  __shared__ __align__(16) unsigned short s_wab[MAB * XSTR];    // stacked a_w|b_w bf16
  __shared__ __align__(16) unsigned short s_g[C_OUT * GSTR];    // stacked g_w bf16 [o][i*64+c]
  __shared__ float s_biasab[MAB];
  __shared__ float s_bnA[C_OUT];
  __shared__ float s_bnC[C_OUT];
  __shared__ float s_colsum[NSUB * V_DIM + 1];

  const int tid  = threadIdx.x;
  const int lane = tid & 31;
  const int wave = tid >> 5;
  const int n    = blockIdx.y;
  const int t0   = blockIdx.x * TT;
  const int Tta  = min(TT, T_DIM - t0);
  const int Pv   = Tta * V_DIM;

  const float* xg = x   + (size_t)n * C_IN  * HW + t0 * V_DIM;
  float*       og = out + (size_t)n * C_OUT * HW + t0 * V_DIM;

  // ---------------- phase A: async-stage x tile; load weights ----------------
  float* staged = s_abxs;                       // f32 [c][p], 64*208
  for (int e = tid; e < C_IN * PPAD; e += 256) {
    int p = e % PPAD, c = e / PPAD;
    if (p < Pv)
      async_load_f32_to_lds(&staged[c * PPAD + p], xg + c * HW + p);
  }
  for (int e = tid; e < MAB * C_IN; e += 256) {
    int row = e / C_IN, c = e % C_IN;
    float v = (row < 48) ? a_w[row * C_IN + c] : b_w[(row - 48) * C_IN + c];
    s_wab[row * XSTR + c] = f2bf(v);
  }
  if (tid < MAB) s_biasab[tid] = (tid < 48) ? a_b[tid] : b_b[tid - 48];
  for (int e = tid; e < C_OUT * GK; e += 256) {
    int o = e / GK, k = e % GK;
    int i = k >> 6, c = k & 63;
    s_g[o * GSTR + k] = f2bf(g_w[((i << 6) + o) * C_IN + c]);
  }
  if (tid < C_OUT) {
    float inv = rsqrtf(bn_var[tid] + 1e-5f);
    float a   = bn_gamma[tid] * inv;
    float gb  = g_b[tid] + g_b[64 + tid] + g_b[128 + tid];
    s_bnA[tid] = a;
    s_bnC[tid] = gb * a + bn_beta[tid] - bn_mean[tid] * a;
  }
  if (tid < NSUB * V_DIM) {
    int i = tid / V_DIM, v = tid % V_DIM;
    float s = 0.f;
    for (int u = 0; u < V_DIM; ++u) {
      int idx = (i * V_DIM + u) * V_DIM + v;
      s += Amat[idx] + gattn[idx];
    }
    s_colsum[tid] = s;
  }
  async_wait_all();
  __syncthreads();

  // ---------------- phase B: convert staged f32 -> bf16 [p][c] ----------------
  for (int e = tid; e < C_IN * PPAD; e += 256) {
    int p = e % PPAD, c = e / PPAD;
    float v = (p < Pv) ? staged[c * PPAD + p] : 0.f;
    s_xbf[p * XSTR + c] = f2bf(v);
  }
  __syncthreads();

  // init scale with column-sums of (A + graph_attn)
  for (int e = tid; e < NSUB * PPAD; e += 256) {
    int i = e / PPAD, p = e % PPAD;
    s_scale[e] = (p < Pv) ? s_colsum[i * V_DIM + p % V_DIM] : 0.f;
  }

  // ---------------- GEMM1: AB(96 x P) = Wab(96x64) x Xbf(64 x P) + bias ----------------
  for (int tile = wave; tile < 6 * 13; tile += 8) {
    int mt = tile % 6, nt = tile / 6;
    int arow = mt * 16 + (lane & 15);
    int bcol = nt * 16 + (lane & 15);
    v8f acc = {};
#pragma unroll
    for (int kk = 0; kk < 2; ++kk) {
      v16bf a = load_fragA(s_wab, arow, kk * 32, XSTR, lane);
      v16bf b = load_fragB(s_xbf, bcol, kk * 32, XSTR, lane);
      acc = __builtin_amdgcn_wmma_f32_16x16x32_bf16(false, a, false, b, (short)0,
                                                    acc, false, false);
    }
    int hi8 = (lane & 16) ? 8 : 0;
    int col = nt * 16 + (lane & 15);
#pragma unroll
    for (int r = 0; r < 8; ++r) {
      int row = mt * 16 + r + hi8;
      s_abxs[row * PPAD + col] = acc[r] + s_biasab[row];
    }
  }
  __syncthreads();

  // ---------------- attention: S row, softmax over w, accumulate scale ----------------
  const float* AB = s_abxs;
  for (int R = tid; R < NSUB * Tta * V_DIM; R += 256) {
    int i   = R / (Tta * V_DIM);
    int rem = R % (Tta * V_DIM);
    int tl  = rem / V_DIM, u = rem % V_DIM;
    int pu  = tl * V_DIM + u;
    float a1[INTER_C];
#pragma unroll
    for (int cc = 0; cc < INTER_C; ++cc)
      a1[cc] = AB[(i * 16 + cc) * PPAD + pu];
    float lg[V_DIM];
    float mx = -1e30f;
    for (int w = 0; w < V_DIM; ++w) {
      const float* a2 = AB + (48 + i * 16) * PPAD + tl * V_DIM + w;
      float s = 0.f;
#pragma unroll
      for (int cc = 0; cc < INTER_C; ++cc)
        s += a1[cc] * a2[cc * PPAD];
      s *= (1.f / INTER_C);
      lg[w] = s;
      mx = fmaxf(mx, s);
    }
    float sum = 0.f;
    for (int w = 0; w < V_DIM; ++w) { float e = __expf(lg[w] - mx); lg[w] = e; sum += e; }
    float inv = 1.f / sum;
    float* sc = s_scale + i * PPAD + tl * V_DIM;
    for (int w = 0; w < V_DIM; ++w)
      atomicAdd(&sc[w], lg[w] * inv);          // ds_add_f32
  }
  __syncthreads();

  // ---------------- build xs = stacked (x .* scale_i), bf16, reusing AB storage ----------------
  unsigned short* s_xs = (unsigned short*)s_abxs;
  for (int e = tid; e < PPAD * GK; e += 256) {
    int p = e / GK, k = e % GK;
    int i = k >> 6, c = k & 63;
    float v = 0.f;
    if (p < Pv) v = bf2f(s_xbf[p * XSTR + c]) * s_scale[i * PPAD + p];
    s_xs[p * GSTR + k] = f2bf(v);
  }
  __syncthreads();

  // ---------------- GEMM2: hidden(64 x P) = G(64x192) x xs(192 x P); fused BN+res+ReLU ----------------
  for (int tile = wave; tile < 4 * 13; tile += 8) {
    int mt = tile % 4, nt = tile / 4;
    int arow = mt * 16 + (lane & 15);
    int bcol = nt * 16 + (lane & 15);
    v8f acc = {};
#pragma unroll
    for (int kk = 0; kk < 6; ++kk) {
      v16bf a = load_fragA(s_g,  arow, kk * 32, GSTR, lane);
      v16bf b = load_fragB(s_xs, bcol, kk * 32, GSTR, lane);
      acc = __builtin_amdgcn_wmma_f32_16x16x32_bf16(false, a, false, b, (short)0,
                                                    acc, false, false);
    }
    int hi8 = (lane & 16) ? 8 : 0;
    int p   = nt * 16 + (lane & 15);
    if (p < Pv) {
#pragma unroll
      for (int r = 0; r < 8; ++r) {
        int o = mt * 16 + r + hi8;
        float h = acc[r] * s_bnA[o] + s_bnC[o] + xg[o * HW + p];
        og[o * HW + p] = fmaxf(h, 0.f);
      }
    }
  }
}

extern "C" void kernel_launch(void* const* d_in, const int* in_sizes, int n_in,
                              void* d_out, int out_size, void* d_ws, size_t ws_size,
                              hipStream_t stream) {
  (void)in_sizes; (void)n_in; (void)d_ws; (void)ws_size; (void)out_size;
  const float* x        = (const float*)d_in[0];
  const float* A        = (const float*)d_in[1];
  const float* gattn    = (const float*)d_in[2];
  const float* g_w      = (const float*)d_in[3];
  const float* g_b      = (const float*)d_in[4];
  const float* a_w      = (const float*)d_in[5];
  const float* a_b      = (const float*)d_in[6];
  const float* b_w      = (const float*)d_in[7];
  const float* b_b      = (const float*)d_in[8];
  const float* bn_gamma = (const float*)d_in[9];
  const float* bn_beta  = (const float*)d_in[10];
  const float* bn_mean  = (const float*)d_in[11];
  const float* bn_var   = (const float*)d_in[12];
  float* out = (float*)d_out;

  dim3 grid((T_DIM + TT - 1) / TT, NB);
  agc_fused_kernel<<<grid, 256, 0, stream>>>(x, A, gattn, g_w, g_b, a_w, a_b,
                                             b_w, b_b, bn_gamma, bn_beta,
                                             bn_mean, bn_var, out);
}
